// Head_50285477101768
// MI455X (gfx1250) — matrix operand order repaired and checked by reading, not compiled, not run on hardware
//
#include <hip/hip_runtime.h>

// ---------------------------------------------------------------------------
// Causal single-head attention for MI455X (gfx1250), wave32 + WMMA bf16.
//   q,k,v = x@Wq, x@Wk, x@Wv ; out = softmax(mask(q k^T * sqrt(H))) @ v
// B=8 T=2048 D=2048 H=128.
// ---------------------------------------------------------------------------

typedef __attribute__((ext_vector_type(16))) __bf16 v16bf;
typedef __attribute__((ext_vector_type(2)))  __bf16 v2bf;
typedef __attribute__((ext_vector_type(8)))  float  v8f;

#define BATCH  8
#define TSEQ   2048
#define DMODEL 2048
#define HSZ    128
#define MROWS  (BATCH * TSEQ)          // 16384
#define SCALE  11.313708498984761f     // sqrt(128) (faithful to source)
#define NEGV   (-1e9f)

__device__ __forceinline__ unsigned short f2bf(float f) {
  unsigned u = __float_as_uint(f);
  unsigned r = u + 0x7FFFu + ((u >> 16) & 1u);   // round-to-nearest-even
  return (unsigned short)(r >> 16);
}

__device__ __forceinline__ void sched_fence() {
#if __has_builtin(__builtin_amdgcn_sched_barrier)
  __builtin_amdgcn_sched_barrier(0);   // keep issued load groups above compute
#endif
}

union FragBF {                 // one WMMA bf16 A/B operand (16 bf16 per lane)
  v16bf v;
  uint4 u[2];
  unsigned w[8];
  unsigned short s[16];
};
union AccF {                   // one 16x16 f32 C/D tile (8 f32 per lane)
  v8f v;
  float f[8];
};
union Pack8 { unsigned short s[8]; uint4 u; };

__device__ __forceinline__ v8f wmma_bf16(v16bf a, v16bf b, v8f c) {
  // D = A(16x32) * B(32x16) + C ; emits v_wmma_f32_16x16x32_bf16
  return __builtin_amdgcn_wmma_f32_16x16x32_bf16(
      /*neg_a=*/false, a, /*neg_b=*/false, b,
      /*c_mod=*/(short)0, c, /*reuse_a=*/false, /*reuse_b=*/false);
}

// ---------------------------------------------------------------------------
// Kernel 1: W[D,H] f32 -> bf16 transposed [H,D] (B-fragment friendly layout).
// ---------------------------------------------------------------------------
__global__ void convert_w_kernel(const float* __restrict__ Wq,
                                 const float* __restrict__ Wk,
                                 const float* __restrict__ Wv,
                                 unsigned short* __restrict__ wbT) {
  const int p = blockIdx.y;
  const float* W = (p == 0) ? Wq : ((p == 1) ? Wk : Wv);
  int i = blockIdx.x * blockDim.x + threadIdx.x;   // over D*H
  if (i >= DMODEL * HSZ) return;
  int d = i / HSZ, h = i % HSZ;
  wbT[(size_t)p * HSZ * DMODEL + (size_t)h * DMODEL + d] =
      f2bf(W[(size_t)d * HSZ + h]);
}

// ---------------------------------------------------------------------------
// Kernel 2: projections. Each wave: 16 rows x 128 cols of one projection.
//   Software-pipelined K loop (unroll-by-2, double-buffered B fragments and
//   A f32 tiles) so one phase's loads are in flight under the other's WMMAs.
//   p==0 -> q bf16 [T,H]; p==1 -> k bf16 [T,H]; p==2 -> v bf16 transposed [H,T]
// ---------------------------------------------------------------------------
__global__ __launch_bounds__(128)
void proj_kernel(const float* __restrict__ x,
                 const unsigned short* __restrict__ wbT,
                 unsigned short* __restrict__ qb,
                 unsigned short* __restrict__ kb,
                 unsigned short* __restrict__ vT) {
  const int lane = threadIdx.x & 31;
  const int wave = threadIdx.x >> 5;
  const int p    = blockIdx.y;
  const int rowbase = (blockIdx.x * 4 + wave) * 16;   // gridDim.x = MROWS/64
  const int n  = lane & 15;
  const int hi = lane >> 4;

  const unsigned short* Wt = wbT + (size_t)p * HSZ * DMODEL;
  const float* xrow = x + (size_t)(rowbase + n) * DMODEL;  // A: m = lane%16

  AccF acc[8];
#pragma unroll
  for (int t = 0; t < 8; ++t) acc[t].v = (v8f)(0.0f);

  auto loadA = [&](int kc, float4* A) {
    const float4* ap  = (const float4*)(xrow + kc + hi * 8);       // K=hi*8+..
    const float4* ap2 = (const float4*)(xrow + kc + 16 + hi * 8);  // +16
    A[0] = ap[0]; A[1] = ap[1]; A[2] = ap2[0]; A[3] = ap2[1];
  };
  auto loadB = [&](int kc, FragBF* bf) {
#pragma unroll
    for (int nt = 0; nt < 8; ++nt) {
      const uint4* bp =
          (const uint4*)(Wt + (size_t)(nt * 16 + n) * DMODEL + kc + hi * 8);
      bf[nt].u[0] = bp[0];
      bf[nt].u[1] = bp[2];
    }
  };
  auto packA = [&](const float4* A) {
    FragBF a;
#pragma unroll
    for (int j = 0; j < 4; ++j) {
      a.s[4 * j + 0] = f2bf(A[j].x);
      a.s[4 * j + 1] = f2bf(A[j].y);
      a.s[4 * j + 2] = f2bf(A[j].z);
      a.s[4 * j + 3] = f2bf(A[j].w);
    }
    return a;
  };

  float4 A0[4], A1[4];
  FragBF B0[8], B1[8];
  loadA(0, A0);
  loadB(0, B0);
  sched_fence();

  for (int kc = 0; kc < DMODEL; kc += 64) {
    // phase 0: prefetch phase-1 data, compute with phase-0 buffers
    loadA(kc + 32, A1);
    loadB(kc + 32, B1);
    sched_fence();
    FragBF a = packA(A0);
#pragma unroll
    for (int nt = 0; nt < 8; ++nt)
      acc[nt].v = wmma_bf16(a.v, B0[nt].v, acc[nt].v);

    // phase 1: prefetch next iteration's phase-0 data, compute phase-1
    if (kc + 64 < DMODEL) {
      loadA(kc + 64, A0);
      loadB(kc + 64, B0);
      sched_fence();
    }
    a = packA(A1);
#pragma unroll
    for (int nt = 0; nt < 8; ++nt)
      acc[nt].v = wmma_bf16(a.v, B1[nt].v, acc[nt].v);
  }

  if (p < 2) {                                    // q / k: natural [T,H] bf16
    unsigned short* out = (p == 0) ? qb : kb;
#pragma unroll
    for (int nt = 0; nt < 8; ++nt)
#pragma unroll
      for (int r = 0; r < 8; ++r) {
        int m = r + hi * 8;                       // C layout row
        out[(size_t)(rowbase + m) * HSZ + nt * 16 + n] = f2bf(acc[nt].f[r]);
      }
  } else {                                        // v: transposed [H,T] bf16
    const int b  = rowbase / TSEQ;
    const int t0 = rowbase % TSEQ;
#pragma unroll
    for (int nt = 0; nt < 8; ++nt) {
      const int h = nt * 16 + n;
      Pack8 pk;
#pragma unroll
      for (int r = 0; r < 8; ++r) pk.s[r] = f2bf(acc[nt].f[r]); // t=t0+hi*8+r
      *(uint4*)(vT + (size_t)b * HSZ * TSEQ + (size_t)h * TSEQ + t0 + hi * 8) =
          pk.u;
    }
  }
}

// ---------------------------------------------------------------------------
// Kernel 3: flash attention. One wave per (batch, 16-query tile).
//   V fragments are issued at the top of each key chunk; their latency is
//   hidden behind the S WMMAs and the whole softmax VALU section.
// ---------------------------------------------------------------------------
__global__ __launch_bounds__(128)
void attn_kernel(const unsigned short* __restrict__ qb,
                 const unsigned short* __restrict__ kb,
                 const unsigned short* __restrict__ vT,
                 float* __restrict__ out) {
  __shared__ __align__(16) unsigned short pl[4][16 * 32];  // per-wave P tile

  const int lane = threadIdx.x & 31;
  const int wave = threadIdx.x >> 5;
  const int n  = lane & 15;
  const int hi = lane >> 4;

  const int tileIdx = blockIdx.x * 4 + wave;   // gridDim.x = B*(T/16)/4
  const int b  = tileIdx >> 7;                 // / (T/16 = 128)
  const int qt = tileIdx & 127;
  const int qbase = qt * 16;

  const unsigned short* qB = qb + (size_t)b * TSEQ * HSZ;
  const unsigned short* kB = kb + (size_t)b * TSEQ * HSZ;
  const unsigned short* vB = vT + (size_t)b * HSZ * TSEQ;

  // Resident Q fragments: A-layout over the H (=GEMM-K) dimension.
  FragBF qf[4];
  {
    const unsigned short* qrow = qB + (size_t)(qbase + n) * HSZ;
#pragma unroll
    for (int hc = 0; hc < 4; ++hc) {
      qf[hc].u[0] = *(const uint4*)(qrow + hc * 32 + hi * 16);
      qf[hc].u[1] = *(const uint4*)(qrow + hc * 32 + hi * 16 + 8);
    }
  }

  AccF oacc[8];
#pragma unroll
  for (int t = 0; t < 8; ++t) oacc[t].v = (v8f)(0.0f);
  float mrun[8], lrun[8];
#pragma unroll
  for (int r = 0; r < 8; ++r) { mrun[r] = -__builtin_inff(); lrun[r] = 0.0f; }

  const int nkc = ((qbase + 15) >> 5) + 1;     // causal: 32-key chunks needed
  unsigned short* myP = pl[wave];

  for (int kc = 0; kc < nkc; ++kc) {
    const int kbase = kc * 32;

    // ---- issue ALL loads for this chunk: K frags then V frags ----
    FragBF kf0[4], kf1[4], vf[8];
    {
      const unsigned short* k0 = kB + (size_t)(kbase + n) * HSZ;
      const unsigned short* k1 = kB + (size_t)(kbase + 16 + n) * HSZ;
#pragma unroll
      for (int hc = 0; hc < 4; ++hc) {
        kf0[hc].u[0] = *(const uint4*)(k0 + hc * 32 + hi * 16);
        kf0[hc].u[1] = *(const uint4*)(k0 + hc * 32 + hi * 16 + 8);
        kf1[hc].u[0] = *(const uint4*)(k1 + hc * 32 + hi * 16);
        kf1[hc].u[1] = *(const uint4*)(k1 + hc * 32 + hi * 16 + 8);
      }
#pragma unroll
      for (int ht = 0; ht < 8; ++ht) {
        const unsigned short* vp =
            vB + (size_t)(ht * 16 + n) * TSEQ + kbase + hi * 16;
        vf[ht].u[0] = *(const uint4*)(vp);
        vf[ht].u[1] = *(const uint4*)(vp + 8);
      }
      sched_fence();
    }

    // ---- S = q @ k^T for 32 keys (two 16x16 tiles, 4 chained WMMA each) ----
    AccF s0, s1;
    s0.v = (v8f)(0.0f);
    s1.v = (v8f)(0.0f);
#pragma unroll
    for (int hc = 0; hc < 4; ++hc) {
      s0.v = wmma_bf16(qf[hc].v, kf0[hc].v, s0.v);
      s1.v = wmma_bf16(qf[hc].v, kf1[hc].v, s1.v);
    }

    // ---- scale, causal mask, online-softmax update ----
    float p0[8], p1[8];
#pragma unroll
    for (int r = 0; r < 8; ++r) {
      const int gq = qbase + r + hi * 8;        // global query row
      float v0 = ((kbase + n)      <= gq) ? s0.f[r] * SCALE : NEGV;
      float v1 = ((kbase + 16 + n) <= gq) ? s1.f[r] * SCALE : NEGV;
      float rm = fmaxf(v0, v1);
#pragma unroll
      for (int off = 8; off >= 1; off >>= 1)     // row max across 16 cols
        rm = fmaxf(rm, __shfl_xor(rm, off, 32));
      const float nm = fmaxf(mrun[r], rm);
      const float e0 = __expf(v0 - nm);
      const float e1 = __expf(v1 - nm);
      float rs = e0 + e1;
#pragma unroll
      for (int off = 8; off >= 1; off >>= 1)     // row sum across 16 cols
        rs += __shfl_xor(rs, off, 32);
      const float corr = __expf(mrun[r] - nm);
      mrun[r] = nm;
      lrun[r] = lrun[r] * corr + rs;
      p0[r] = e0;
      p1[r] = e1;
#pragma unroll
      for (int t = 0; t < 8; ++t) oacc[t].f[r] *= corr;
    }

    // ---- C-layout -> A-layout for P via per-wave LDS tile ----
#pragma unroll
    for (int r = 0; r < 8; ++r) {
      const int m = r + hi * 8;
      myP[m * 32 + n]      = f2bf(p0[r]);
      myP[m * 32 + 16 + n] = f2bf(p1[r]);
    }
    asm volatile("s_wait_dscnt 0" ::: "memory"); // wave-local RAW through LDS

    FragBF pf;
    {
      const uint4* lp = (const uint4*)myP;       // A: m = lane%16 = n
      pf.u[0] = lp[n * 4 + hi * 2];              // K = hi*16 + 0..7
      pf.u[1] = lp[n * 4 + hi * 2 + 1];          // K = hi*16 + 8..15
    }

    // ---- out += P(16x32) @ V(32x128): V already resident ----
#pragma unroll
    for (int ht = 0; ht < 8; ++ht)
      oacc[ht].v = wmma_bf16(pf.v, vf[ht].v, oacc[ht].v);
  }

  // ---- epilogue: divide by row sum, store f32 [B,T,H] ----
  float* orow = out + (size_t)b * TSEQ * HSZ + (size_t)qbase * HSZ;
#pragma unroll
  for (int r = 0; r < 8; ++r) {
    const int m = r + hi * 8;
    const float inv = 1.0f / lrun[r];
#pragma unroll
    for (int ht = 0; ht < 8; ++ht)
      orow[(size_t)m * HSZ + ht * 16 + n] = oacc[ht].f[r] * inv;
  }
}

// ---------------------------------------------------------------------------
extern "C" void kernel_launch(void* const* d_in, const int* in_sizes, int n_in,
                              void* d_out, int out_size, void* d_ws,
                              size_t ws_size, hipStream_t stream) {
  (void)in_sizes; (void)n_in; (void)out_size; (void)ws_size;
  const float* x  = (const float*)d_in[0];
  const float* Wq = (const float*)d_in[1];
  const float* Wk = (const float*)d_in[2];
  const float* Wv = (const float*)d_in[3];
  float* out = (float*)d_out;

  // Workspace layout (bf16), ~13.6 MB total:
  unsigned short* wbT = (unsigned short*)d_ws;              // [3][H][D]
  unsigned short* qb  = wbT + (size_t)3 * HSZ * DMODEL;     // [B*T][H]
  unsigned short* kb  = qb  + (size_t)MROWS * HSZ;          // [B*T][H]
  unsigned short* vT  = kb  + (size_t)MROWS * HSZ;          // [B][H][T]

  dim3 gw((DMODEL * HSZ + 255) / 256, 3);
  convert_w_kernel<<<gw, 256, 0, stream>>>(Wq, Wk, Wv, wbT);

  dim3 gp(MROWS / 64, 3);                                   // 4 waves/block
  proj_kernel<<<gp, 128, 0, stream>>>(x, wbT, qb, kb, vT);

  dim3 ga((BATCH * (TSEQ / 16)) / 4);                       // 256 blocks
  attn_kernel<<<ga, 128, 0, stream>>>(qb, kb, vT, out);
}
